// InnTransBlock_4252017623229
// MI455X (gfx1250) — matrix operand order repaired and checked by reading, not compiled, not run on hardware
//
#include <hip/hip_runtime.h>
#include <math.h>

#define HW_   (256 * 256)
#define IMG   256
#define BATCH 2

typedef __attribute__((ext_vector_type(16))) _Float16     v16h;
typedef __attribute__((ext_vector_type(8)))  float        v8f;
typedef __attribute__((ext_vector_type(4)))  unsigned int u32x4;
typedef __attribute__((ext_vector_type(8)))  int          i32x8;
typedef __attribute__((ext_vector_type(4)))  int          i32x4;

#if defined(__HIP_DEVICE_COMPILE__)
#if __has_builtin(__builtin_amdgcn_tensor_load_to_lds)
#define HAVE_TDM 1
#endif
#endif

// ---------------------------------------------------------------------------
// Fragment index helpers for 16-bit WMMA operands (wave32, 16x16x32):
//  A (16xK tile row-major, M = out-channel): lane m=lane&15, K chunks
//  {0..15} for lanes 0-15, {16..31} for lanes 16-31 (K = e + 16*(lane>>4)).
//  B (Kx16, N = pixel): lane n=lane&15, element e -> K = e + 16*(lane>>4).
//  C/D f32: lane&15 = N col, VGPR r -> M = r + 8*(lane>>4).
// ---------------------------------------------------------------------------

// Pointwise (1x1-conv / linear) GEMM over pixels with fused epilogues.
// Y[o,p] = epi( sum_k W[o,k] * X[k,p] ).  Weights staged to LDS via the
// Tensor Data Mover; each wave computes 4 pixel tiles (64 px) x 16 out-ch,
// so one A (weight) fragment feeds 4 back-to-back WMMAs per K-step.
// mode: 1=BN  2=BN+ReLU6  3=+bias  4=+bias,GELU  5=+bias,+residual
template <int K>
__global__ void __launch_bounds__(256)
gemm_pw(const float* __restrict__ X, const float* __restrict__ W,
        float* __restrict__ Y,
        const float* __restrict__ bn_g, const float* __restrict__ bn_b,
        const float* __restrict__ bn_m, const float* __restrict__ bn_v,
        const float* __restrict__ bias, const float* __restrict__ resid,
        int ldw, int wtrans, long xB, long yB, long rB, int mode)
{
  __shared__ float lw[K * 16];
  const int o0 = blockIdx.y * 16;

#if defined(HAVE_TDM)
  if (threadIdx.x < 32) {
    // Tensor DMA Descriptor (ISA 8.3/8.4): one 2D fp32 tile of the weight
    // matrix -> compacted row-major in LDS.
    const float* wsrc = wtrans ? (W + o0) : (W + (long)o0 * ldw);
    unsigned long long ga = (unsigned long long)(size_t)wsrc;
    unsigned ldsoff = (unsigned)(size_t)(&lw[0]);       // low 32 bits = LDS offset
    unsigned tile_d0 = wtrans ? 16u : (unsigned)K;      // contiguous dim
    unsigned tile_d1 = wtrans ? (unsigned)K : 16u;      // rows
    unsigned td0 = (unsigned)ldw;                       // tensor dim0 (full row)
    unsigned td1 = tile_d1;
    unsigned long long st0 = (unsigned long long)ldw;   // dim0 stride (elements)
    u32x4 g0;
    g0[0] = 1u;                                         // count=1, user mode
    g0[1] = ldsoff;                                     // lds_addr
    g0[2] = (unsigned)(ga & 0xFFFFFFFFull);             // global_addr lo
    g0[3] = (unsigned)((ga >> 32) & 0x1FFFFFFull) | (2u << 30); // hi | type=2
    i32x8 g1;
    g1[0] = (int)(2u << 16);                            // data_size=4B
    g1[1] = (int)((td0 & 0xFFFFu) << 16);               // tensor_dim0[15:0]
    g1[2] = (int)(((td0 >> 16) & 0xFFFFu) | ((td1 & 0xFFFFu) << 16));
    g1[3] = (int)(((td1 >> 16) & 0xFFFFu) | (tile_d0 << 16)); // tile_dim0
    g1[4] = (int)(tile_d1 & 0xFFFFu);                   // tile_dim1, tile_dim2=0
    g1[5] = (int)(st0 & 0xFFFFFFFFull);                 // dim0_stride lo
    g1[6] = (int)((st0 >> 32) & 0xFFFFull);             // dim0_stride hi
    g1[7] = 0;
    i32x4 gz = {0, 0, 0, 0};
#if __clang_major__ >= 23
    i32x8 gz8 = {0, 0, 0, 0, 0, 0, 0, 0};
    __builtin_amdgcn_tensor_load_to_lds(g0, g1, gz, gz, gz8, 0);
#else
    __builtin_amdgcn_tensor_load_to_lds(g0, g1, gz, gz, 0);
#endif
    __builtin_amdgcn_s_wait_tensorcnt(0);
  }
#else
  for (int i = threadIdx.x; i < 16 * K; i += 256) {
    if (!wtrans) { int o = i / K,  k = i - o * K;  lw[i] = W[(long)(o0 + o) * ldw + k]; }
    else         { int k = i / 16, o = i - k * 16; lw[i] = W[(long)k * ldw + o0 + o]; }
  }
#endif
  __syncthreads();

  const int lane = threadIdx.x & 31;
  const int wid  = threadIdx.x >> 5;
  const int m    = lane & 15;
  const int hi   = lane >> 4;
  const int p0   = (blockIdx.x * 32 + wid * 4) * 16;   // 4 pixel tiles per wave
  const float* Xb = X + (long)blockIdx.z * xB;

  v8f acc0 = {}, acc1 = {}, acc2 = {}, acc3 = {};
#pragma unroll
  for (int k0 = 0; k0 < K; k0 += 32) {
    if (k0 + 32 < K) {
      __builtin_prefetch(Xb + (long)(k0 + 32) * HW_ + p0,      0, 1);
      __builtin_prefetch(Xb + (long)(k0 + 32) * HW_ + p0 + 32, 0, 1);
    }
    v16h a;
#pragma unroll
    for (int e = 0; e < 16; ++e) {
      int ka = k0 + hi * 16 + e;
      a[e] = (_Float16)(wtrans ? lw[ka * 16 + m] : lw[m * K + ka]);
    }
    v16h b0, b1, b2, b3;
#pragma unroll
    for (int e = 0; e < 16; ++e) {
      long rowoff = (long)(k0 + hi * 16 + e) * HW_ + p0 + m;
      b0[e] = (_Float16)Xb[rowoff];
      b1[e] = (_Float16)Xb[rowoff + 16];
      b2[e] = (_Float16)Xb[rowoff + 32];
      b3[e] = (_Float16)Xb[rowoff + 48];
    }
    acc0 = __builtin_amdgcn_wmma_f32_16x16x32_f16(false, a, false, b0, (short)0, acc0, false, false);
    acc1 = __builtin_amdgcn_wmma_f32_16x16x32_f16(false, a, false, b1, (short)0, acc1, false, false);
    acc2 = __builtin_amdgcn_wmma_f32_16x16x32_f16(false, a, false, b2, (short)0, acc2, false, false);
    acc3 = __builtin_amdgcn_wmma_f32_16x16x32_f16(false, a, false, b3, (short)0, acc3, false, false);
  }

  const long yb = (long)blockIdx.z * yB;
#pragma unroll
  for (int t = 0; t < 4; ++t) {
    const v8f acc = (t == 0) ? acc0 : (t == 1) ? acc1 : (t == 2) ? acc2 : acc3;
    const int p = p0 + t * 16 + m;
#pragma unroll
    for (int r = 0; r < 8; ++r) {
      int o = o0 + r + hi * 8;
      float val = acc[r];
      if (mode == 1 || mode == 2) {
        float sc = bn_g[o] * rsqrtf(bn_v[o] + 1e-5f);
        val = (val - bn_m[o]) * sc + bn_b[o];
        if (mode == 2) val = fminf(fmaxf(val, 0.f), 6.f);
      } else if (mode >= 3) {
        val += bias[o];
        if (mode == 4) val = 0.5f * val * (1.f + erff(val * 0.70710678118654752f));
        if (mode == 5) val += resid[(long)blockIdx.z * rB + (long)o * HW_ + p];
      }
      Y[yb + (long)o * HW_ + p] = val;
    }
  }
}

// Depthwise 3x3 conv, reflect pad 1, fused BN + ReLU6.
__global__ void __launch_bounds__(256)
dw3x3_bn(const float* __restrict__ X, const float* __restrict__ Wd,
         const float* __restrict__ g, const float* __restrict__ b,
         const float* __restrict__ m, const float* __restrict__ v,
         float* __restrict__ Y, long xB, long yB)
{
  int p = blockIdx.x * 256 + threadIdx.x;
  int c = blockIdx.y, bz = blockIdx.z;
  int py = p >> 8, px = p & 255;
  const float* Xc = X + (long)bz * xB + (long)c * HW_;
  float acc = 0.f;
#pragma unroll
  for (int dy = -1; dy <= 1; ++dy) {
    int yy = py + dy; yy = (yy < 0) ? 1 : ((yy >= IMG) ? IMG - 2 : yy);
#pragma unroll
    for (int dx = -1; dx <= 1; ++dx) {
      int xx = px + dx; xx = (xx < 0) ? 1 : ((xx >= IMG) ? IMG - 2 : xx);
      acc += Xc[yy * IMG + xx] * Wd[c * 9 + (dy + 1) * 3 + (dx + 1)];
    }
  }
  float sc = g[c] * rsqrtf(v[c] + 1e-5f);
  float val = (acc - m[c]) * sc + b[c];
  Y[(long)bz * yB + (long)c * HW_ + p] = fminf(fmaxf(val, 0.f), 6.f);
}

// y = x * exp(2*sigmoid(g)-1) + h   (CLAMP = 1.0), over 32 channels.
__global__ void __launch_bounds__(256)
coupling_combine(const float* __restrict__ xi, const float* __restrict__ go,
                 const float* __restrict__ ho, float* __restrict__ yo,
                 long xB, long gB, long yB)
{
  long i  = (long)blockIdx.x * 256 + threadIdx.x; // over 32*HW
  long bz = blockIdx.z;
  float gv = go[bz * gB + i];
  float s  = 1.f / (1.f + expf(-gv));
  yo[bz * yB + i] = xi[bz * xB + i] * expf(2.f * s - 1.f) + ho[bz * gB + i];
}

// y = res + LN(h) over C=96 channels (channel-major planes).
__global__ void __launch_bounds__(256)
ln_residual(const float* __restrict__ h, const float* __restrict__ res,
            const float* __restrict__ g, const float* __restrict__ b,
            float* __restrict__ y, long hB, long rB, long yB)
{
  int p = blockIdx.x * 256 + threadIdx.x;
  long hb = (long)blockIdx.z * hB;
  float mu = 0.f, s2 = 0.f;
  for (int c = 0; c < 96; ++c) {
    float t = h[hb + (long)c * HW_ + p];
    mu += t; s2 += t * t;
  }
  mu *= (1.f / 96.f);
  float var = s2 * (1.f / 96.f) - mu * mu;
  float inv = rsqrtf(var + 1e-5f);
  for (int c = 0; c < 96; ++c) {
    float t = (h[hb + (long)c * HW_ + p] - mu) * inv * g[c] + b[c];
    y[(long)blockIdx.z * yB + (long)c * HW_ + p] =
        res[(long)blockIdx.z * rB + (long)c * HW_ + p] + t;
  }
}

// Shifted-window MSA: one block per (window, head, batch). 512 threads.
// Phase 1: sim = QK^T/sqrt(d) + relpos (+shift mask) via 16 WMMA tiles.
// Phase 2: row softmax in LDS.  Phase 3: out = P@V via 8 tiles x 2 K-steps.
__global__ void __launch_bounds__(512)
wmsa_attn(const float* __restrict__ qkv, const float* __restrict__ relpos,
          float* __restrict__ attn)
{
  __shared__ float sim[64][65];
  __shared__ int pix[64];
  const int win = blockIdx.x, head = blockIdx.y, bz = blockIdx.z;
  const int wy = win >> 5, wx = win & 31;
  if (threadIdx.x < 64) {
    int ly = threadIdx.x >> 3, lx = threadIdx.x & 7;
    int gy = (wy * 8 + ly + 4) & 255;   // roll(-shift): rolled[i] = x[i+4]
    int gx = (wx * 8 + lx + 4) & 255;
    pix[threadIdx.x] = gy * IMG + gx;
  }
  __syncthreads();

  const int lane = threadIdx.x & 31;
  const int wid  = threadIdx.x >> 5;
  const int m = lane & 15, hi = lane >> 4;
  const long qb = (long)bz * 288 * HW_;
  const int hc = head * 32;

  { // ---- phase 1: sim tiles ----
    int qi = wid >> 2, kj = wid & 3;
    int qrow = qi * 16 + m, kcol = kj * 16 + m;
    v16h a, bf;
#pragma unroll
    for (int e = 0; e < 16; ++e) {
      int da = hi * 16 + e;
      a[e]  = (_Float16)qkv[qb + (long)(hc + da) * HW_ + pix[qrow]];
      int db = hi * 16 + e;
      bf[e] = (_Float16)qkv[qb + (long)(96 + hc + db) * HW_ + pix[kcol]];
    }
    v8f acc = {};
    acc = __builtin_amdgcn_wmma_f32_16x16x32_f16(false, a, false, bf,
                                                 (short)0, acc, false, false);
#pragma unroll
    for (int r = 0; r < 8; ++r) {
      int q = qi * 16 + r + hi * 8, kk = kj * 16 + m;
      int qy = q >> 3, qx = q & 7, ky = kk >> 3, kx = kk & 7;
      float val = acc[r] * 0.17677669529663687f; // 1/sqrt(32)
      val += relpos[head * 225 + (qy - ky + 7) * 15 + (qx - kx + 7)];
      bool msk = ((wy == 31) && ((qy < 4) != (ky < 4))) ||
                 ((wx == 31) && ((qx < 4) != (kx < 4)));
      sim[q][kk] = msk ? -1e30f : val;
    }
  }
  __syncthreads();

  if (threadIdx.x < 64) { // ---- phase 2: softmax ----
    int q = threadIdx.x;
    float mx = -1e30f;
    for (int j = 0; j < 64; ++j) mx = fmaxf(mx, sim[q][j]);
    float sum = 0.f;
    for (int j = 0; j < 64; ++j) { float e = expf(sim[q][j] - mx); sim[q][j] = e; sum += e; }
    float inv = 1.f / sum;
    for (int j = 0; j < 64; ++j) sim[q][j] *= inv;
  }
  __syncthreads();

  if (wid < 8) { // ---- phase 3: out = P @ V ----
    int mi = wid >> 1, nj = wid & 1;
    v8f acc = {};
#pragma unroll
    for (int k0 = 0; k0 < 64; k0 += 32) {
      v16h a, bf;
      int qrow = mi * 16 + m;
#pragma unroll
      for (int e = 0; e < 16; ++e) {
        int ka = k0 + hi * 16 + e;
        a[e] = (_Float16)sim[qrow][ka];
        int kb = k0 + hi * 16 + e;
        bf[e] = (_Float16)qkv[qb + (long)(192 + hc + nj * 16 + m) * HW_ + pix[kb]];
      }
      acc = __builtin_amdgcn_wmma_f32_16x16x32_f16(false, a, false, bf,
                                                   (short)0, acc, false, false);
    }
    long ab = (long)bz * 96 * HW_;
#pragma unroll
    for (int r = 0; r < 8; ++r) {
      int q = mi * 16 + r + hi * 8;
      attn[ab + (long)(hc + nj * 16 + m) * HW_ + pix[q]] = acc[r];
    }
  }
}

// ---------------------------------------------------------------------------
static void launch_gemm(int K, int OC, const float* X, const float* W, float* Y,
                        const float* g, const float* b, const float* m, const float* v,
                        const float* bias, const float* resid,
                        int ldw, int wtrans, long xB, long yB, long rB, int mode,
                        hipStream_t s)
{
  dim3 grid(HW_ / 512, OC / 16, BATCH);   // 8 waves x 4 pixel tiles x 16 px
  dim3 blk(256);
  switch (K) {
    case 32:  gemm_pw<32 ><<<grid, blk, 0, s>>>(X, W, Y, g, b, m, v, bias, resid, ldw, wtrans, xB, yB, rB, mode); break;
    case 64:  gemm_pw<64 ><<<grid, blk, 0, s>>>(X, W, Y, g, b, m, v, bias, resid, ldw, wtrans, xB, yB, rB, mode); break;
    case 96:  gemm_pw<96 ><<<grid, blk, 0, s>>>(X, W, Y, g, b, m, v, bias, resid, ldw, wtrans, xB, yB, rB, mode); break;
    case 128: gemm_pw<128><<<grid, blk, 0, s>>>(X, W, Y, g, b, m, v, bias, resid, ldw, wtrans, xB, yB, rB, mode); break;
    case 192: gemm_pw<192><<<grid, blk, 0, s>>>(X, W, Y, g, b, m, v, bias, resid, ldw, wtrans, xB, yB, rB, mode); break;
    case 384: gemm_pw<384><<<grid, blk, 0, s>>>(X, W, Y, g, b, m, v, bias, resid, ldw, wtrans, xB, yB, rB, mode); break;
    default: break;
  }
}

extern "C" void kernel_launch(void* const* d_in, const int* in_sizes, int n_in,
                              void* d_out, int out_size, void* d_ws, size_t ws_size,
                              hipStream_t stream)
{
  (void)in_sizes; (void)n_in; (void)out_size; (void)ws_size;
  const float* x = (const float*)d_in[0];
  float* out = (float*)d_out;
  float* ws  = (float*)d_ws;
  auto P = [&](int i) { return (const float*)d_in[i]; };

  const long PB = (long)HW_;        // per-channel plane
  const long XB = 192L * HW_;       // batch stride of x / out

  size_t off = 0;
  auto alloc = [&](int C) { float* p = ws + off; off += (size_t)C * BATCH * HW_; return p; };
  float* ybuf = alloc(192);  // concat([y1,y2,y3, t2]) staged for final conv
  float* bufA = alloc(128);  // bottleneck hidden #1 / reused as attn out
  float* bufB = alloc(128);  // bottleneck hidden #2 / reused as proj/mlp2 out
  float* bufG = alloc(32);
  float* bufH = alloc(32);
  float* bufQ = alloc(384);  // qkv (288) then mlp hidden (384)
  float* bufT = alloc(96);   // t1 = t + LN(attn path)
  float* attnb = bufA;
  float* bufP  = bufB;

  // param index base per bottleneck: +0 w1, +1..4 bn1{g,b,m,v}, +5 wd,
  // +6..9 bn2, +10 w3, +11..14 bn3
  auto bneck = [&](int base, const float* z, long zB, int K, int hid,
                   float* obuf, long oB) {
    launch_gemm(K, hid, z, P(base + 0), bufA, P(base + 1), P(base + 2), P(base + 3),
                P(base + 4), nullptr, nullptr, K, 0, zB, (long)hid * HW_, 0, 2, stream);
    dw3x3_bn<<<dim3(HW_ / 256, hid, BATCH), 256, 0, stream>>>(
        bufA, P(base + 5), P(base + 6), P(base + 7), P(base + 8), P(base + 9),
        bufB, (long)hid * HW_, (long)hid * HW_);
    launch_gemm(hid, 32, bufB, P(base + 10), obuf, P(base + 11), P(base + 12),
                P(base + 13), P(base + 14), nullptr, nullptr, hid, 0,
                (long)hid * HW_, oB, 0, 1, stream);
  };

  // ---- coupling: x1=ch96..127, x2=ch128..159, x3=ch160..191, x4=ch128..191
  bneck(1,  x + 128 * PB, XB, 64, 128, bufG, 32L * HW_);  // G1(x4)
  bneck(16, x + 128 * PB, XB, 64, 128, bufH, 32L * HW_);  // H1(x4)
  coupling_combine<<<dim3(32 * HW_ / 256, 1, BATCH), 256, 0, stream>>>(
      x + 96 * PB, bufG, bufH, ybuf, XB, 32L * HW_, XB);             // y1
  bneck(31, ybuf, XB, 32, 64, bufG, 32L * HW_);           // G2(y1)
  bneck(46, ybuf, XB, 32, 64, bufH, 32L * HW_);           // H2(y1)
  coupling_combine<<<dim3(32 * HW_ / 256, 1, BATCH), 256, 0, stream>>>(
      x + 128 * PB, bufG, bufH, ybuf + 32 * PB, XB, 32L * HW_, XB);  // y2
  bneck(61, ybuf + 32 * PB, XB, 32, 64, bufG, 32L * HW_); // G3(y2)
  bneck(76, ybuf + 32 * PB, XB, 32, 64, bufH, 32L * HW_); // H3(y2)
  coupling_combine<<<dim3(32 * HW_ / 256, 1, BATCH), 256, 0, stream>>>(
      x + 160 * PB, bufG, bufH, ybuf + 64 * PB, XB, 32L * HW_, XB);  // y3

  // ---- transformer on t = ch 0..95
  launch_gemm(96, 288, x, P(91), bufQ, nullptr, nullptr, nullptr, nullptr,
              P(92), nullptr, 288, 1, XB, 288L * HW_, 0, 3, stream);   // qkv
  wmsa_attn<<<dim3(1024, 3, BATCH), 512, 0, stream>>>(bufQ, P(93), attnb);
  launch_gemm(96, 96, attnb, P(94), bufP, nullptr, nullptr, nullptr, nullptr,
              P(95), nullptr, 96, 1, 96L * HW_, 96L * HW_, 0, 3, stream); // proj
  ln_residual<<<dim3(HW_ / 256, 1, BATCH), 256, 0, stream>>>(
      bufP, x, P(96), P(97), bufT, 96L * HW_, XB, 96L * HW_);          // t1
  launch_gemm(96, 384, bufT, P(98), bufQ, nullptr, nullptr, nullptr, nullptr,
              P(99), nullptr, 384, 1, 96L * HW_, 384L * HW_, 0, 4, stream); // mlp1+gelu
  launch_gemm(384, 96, bufQ, P(100), bufP, nullptr, nullptr, nullptr, nullptr,
              P(101), nullptr, 96, 1, 384L * HW_, 96L * HW_, 0, 3, stream); // mlp2
  ln_residual<<<dim3(HW_ / 256, 1, BATCH), 256, 0, stream>>>(
      bufP, bufT, P(102), P(103), ybuf + 96 * PB, 96L * HW_, 96L * HW_, XB); // t2

  // ---- final 1x1 conv over concat + bias + residual -> d_out
  launch_gemm(192, 192, ybuf, P(104), out, nullptr, nullptr, nullptr, nullptr,
              P(105), x, 192, 0, XB, XB, XB, 5, stream);
}
